// Mamba2Simple_41815801594008
// MI455X (gfx1250) — matrix operand order repaired and checked by reading, not compiled, not run on hardware
//
#include <hip/hip_runtime.h>

// ---------------- constants (match reference) ----------------
constexpr int DMODEL  = 256;
constexpr int DSTATE  = 64;
constexpr int HEADDIM = 128;
constexpr int CHUNK   = 128;
constexpr int DINNER  = 512;
constexpr int NH      = 4;
constexpr int CONVDIM = 640;               // DINNER + 2*DSTATE
constexpr int DPROJ   = 1156;              // 2*DINNER + 2*DSTATE + NH
constexpr int NPAD    = 1280;              // DPROJ padded to 10*128
constexpr int BATCH   = 4;
constexpr int HIMG    = 256, WIMG = 512;
constexpr int HC      = 64, WC = 128;      // downsampled spatial
constexpr int L       = HC * WC;           // 8192
constexpr int MTOK    = BATCH * L;         // 32768
constexpr int NC      = L / CHUNK;         // 64 chunks

// ---------------- WMMA fragment helpers ----------------
typedef __attribute__((ext_vector_type(16))) __bf16 v16bf;
typedef __attribute__((ext_vector_type(8)))  float  v8f;
typedef __attribute__((ext_vector_type(4)))  unsigned int v4u;

union FragU { v16bf v; v4u q[2]; };

// A fragment: 16x32 bf16, LDS row-major with leading dim ld.
// lanes 0-15 hold M=lane, K = {0..7, 16..23}; lanes 16-31 hold K = {8..15, 24..31}
__device__ inline v16bf frag_a(const __bf16* base, int ld, int lane) {
  FragU f;
  int m = lane & 15, half = lane >> 4;
  const __bf16* p = base + m * ld + half * 8;
  f.q[0] = *(const v4u*)p;
  f.q[1] = *(const v4u*)(p + 16);
  return f.v;
}

// B fragment: 32x16 bf16, LDS stored [n][k] (K contiguous per column), leading dim ld.
// lanes 0-15: N=lane, K = {0..7, 16..23}; lanes 16-31: N=lane-16, K = {8..15, 24..31}
__device__ inline v16bf frag_b(const __bf16* base, int ld, int lane) {
  FragU f;
  int n = lane & 15, half = lane >> 4;
  const __bf16* p = base + n * ld + half * 8;
  f.q[0] = *(const v4u*)p;
  f.q[1] = *(const v4u*)(p + 16);
  return f.v;
}

__device__ inline v8f wmma_bf16(v16bf a, v16bf b, v8f c) {
  return __builtin_amdgcn_wmma_f32_16x16x32_bf16(false, a, false, b, (short)0, c,
                                                 false, false);
}

// ---------------- CDNA5 async global->LDS copy (ASYNCcnt path) --------------
// ISA: LDS address comes per lane from the VDST VGPR; generic LDS ptr low 32 bits = LDS addr
__device__ inline unsigned lds_off_of(const void* p) {
  return (unsigned)(unsigned long long)p;
}
__device__ inline void async_b128(unsigned lds_off, const void* gptr) {
  asm volatile("global_load_async_to_lds_b128 %0, %1, off"
               :: "v"(lds_off), "v"(gptr) : "memory");
}
__device__ inline void wait_async0() {
  asm volatile("s_wait_asynccnt 0x0" ::: "memory");
}

// ---------------- K1: 4x4/4 downsample conv -> bf16 tokens ----------------
__global__ __launch_bounds__(256)
void downsample_kernel(const float* __restrict__ left, const float* __restrict__ right,
                       const float* __restrict__ w, const float* __restrict__ bias,
                       __bf16* __restrict__ lfbf, __bf16* __restrict__ rfbf) {
  int tok = blockIdx.x;                // 0..MTOK-1
  int img = blockIdx.y;                // 0=left 1=right
  int bi = tok >> 13, l = tok & (L - 1);
  int py = l >> 7, px = l & (WC - 1);
  const float* src = (img ? right : left) + (size_t)bi * 3 * HIMG * WIMG;
  __shared__ float patch[48];
  int t = threadIdx.x;
  if (t < 48) {
    int ic = t >> 4, rem = t & 15, ky = rem >> 2, kx = rem & 3;
    patch[t] = src[((size_t)ic * HIMG + py * 4 + ky) * WIMG + px * 4 + kx];
  }
  __syncthreads();
  float acc = bias[t];
  const float* wr = w + t * 48;
#pragma unroll
  for (int k = 0; k < 48; ++k) acc += wr[k] * patch[k];
  __bf16* o = img ? rfbf : lfbf;
  o[(size_t)tok * DMODEL + t] = (__bf16)acc;
}

// ---------------- K2: weights -> bf16 (in_proj padded, out_proj) -------------
__global__ void pack_weights_kernel(const float* __restrict__ win,
                                    const float* __restrict__ wout,
                                    __bf16* __restrict__ winbf,
                                    __bf16* __restrict__ woutbf) {
  size_t gid = (size_t)blockIdx.x * 256 + threadIdx.x;
  size_t nwin = (size_t)NPAD * DMODEL;
  if (gid < nwin) {
    size_t n = gid / DMODEL, k = gid % DMODEL;
    float v = (n < DPROJ) ? win[n * DMODEL + k] : 0.f;
    winbf[gid] = (__bf16)v;
  }
  if (gid < (size_t)DMODEL * DINNER) woutbf[gid] = (__bf16)wout[gid];
}

// ---------------- generic bf16 GEMM: C[M,N] = A[M,K] * B[N,K]^T --------------
// block = 128(M) x 128(N) tile, 8 waves each computing 32x64 via 2x4 WMMA tiles.
// A and B tiles staged with double-buffered async global->LDS DMA (ASYNCcnt).
__global__ __launch_bounds__(256)
void gemm_tn_kernel(const __bf16* __restrict__ A, int lda,
                    const __bf16* __restrict__ Bw, int ldb,
                    float* __restrict__ C, int ldc, int K, int Nreal) {
  __shared__ __bf16 As[2][128 * 32];
  __shared__ __bf16 Bs[2][128 * 32];
  int t = threadIdx.x, lane = t & 31, wid = t >> 5;
  int m0 = blockIdx.x * 128, n0blk = blockIdx.y * 128;
  int wm = wid >> 1, wn = wid & 1;   // 32-row x 64-col wave region

  int row = t >> 1, cs = (t & 1) * 16;   // 128 rows x 32 cols, 16 elems/thread
  int bn = n0blk + row;
  const __bf16* aptr = A + (size_t)(m0 + row) * lda + cs;
  const __bf16* bptr = Bw + (size_t)bn * ldb + cs;

  auto stage = [&](int buf, int kb) {
    unsigned la = lds_off_of(&As[buf][row * 32 + cs]);
    async_b128(la, aptr + kb);
    async_b128(la + 16, aptr + kb + 8);
    unsigned lb = lds_off_of(&Bs[buf][row * 32 + cs]);
    if (bn < Nreal) {
      async_b128(lb, bptr + kb);
      async_b128(lb + 16, bptr + kb + 8);
    } else {
      v4u z = {0u, 0u, 0u, 0u};
      *(v4u*)&Bs[buf][row * 32 + cs] = z;
      *(v4u*)&Bs[buf][row * 32 + cs + 8] = z;
    }
  };

  v8f acc[2][4] = {};
  stage(0, 0);
  wait_async0();
  __syncthreads();
  for (int kb = 0; kb < K; kb += 32) {
    int cur = (kb >> 5) & 1;
    if (kb + 32 < K) stage(cur ^ 1, kb + 32);   // overlap DMA with WMMA
    const __bf16* Ab = &As[cur][0];
    const __bf16* Bb = &Bs[cur][0];
    v16bf a0 = frag_a(Ab + (wm * 32) * 32, 32, lane);
    v16bf a1 = frag_a(Ab + (wm * 32 + 16) * 32, 32, lane);
#pragma unroll
    for (int j2 = 0; j2 < 4; ++j2) {
      v16bf b = frag_b(Bb + (wn * 64 + j2 * 16) * 32, 32, lane);
      acc[0][j2] = wmma_bf16(a0, b, acc[0][j2]);
      acc[1][j2] = wmma_bf16(a1, b, acc[1][j2]);
    }
    wait_async0();
    __syncthreads();
  }
  for (int i = 0; i < 2; ++i)
    for (int j = 0; j < 4; ++j) {
      int mm = m0 + wm * 32 + i * 16 + (lane >> 4) * 8;
      int n = n0blk + wn * 64 + j * 16 + (lane & 15);
      if (n < Nreal) {
#pragma unroll
        for (int r = 0; r < 8; ++r)
          C[(size_t)(mm + r) * ldc + n] = acc[i][j][r];
      }
    }
}

// ---------------- K4a: dt = softplus(zx[...,1152:] + dt_bias) ----------------
__global__ void dt_kernel(const float* __restrict__ zxl, const float* __restrict__ zxr,
                          const float* __restrict__ dt_bias, float* __restrict__ dt2) {
  size_t gid = (size_t)blockIdx.x * 256 + threadIdx.x;
  if (gid >= (size_t)2 * MTOK * NH) return;
  int h = gid & 3;
  size_t m = (gid >> 2) % MTOK;
  int s = gid >= (size_t)MTOK * NH ? 1 : 0;
  const float* zx = s ? zxr : zxl;
  float x = zx[m * DPROJ + 2 * DINNER + 2 * DSTATE + h] + dt_bias[h];
  dt2[gid] = (x > 20.f) ? x : log1pf(__expf(x));
}

// ---------------- K4b: depthwise causal conv1d + SiLU (left only) ------------
__global__ void conv_silu_kernel(const float* __restrict__ zxl,
                                 const float* __restrict__ conv_w,
                                 const float* __restrict__ conv_b,
                                 float* __restrict__ xconv) {
  int m = blockIdx.x;
  int bi = m >> 13, l = m & (L - 1);
  for (int c = threadIdx.x; c < CONVDIM; c += 256) {
    float acc = conv_b[c];
#pragma unroll
    for (int k = 0; k < 4; ++k) {
      int ls = l - (3 - k);
      if (ls >= 0)
        acc += conv_w[c * 4 + k] * zxl[(size_t)(bi * L + ls) * DPROJ + DINNER + c];
    }
    xconv[(size_t)m * CONVDIM + c] = acc / (1.f + __expf(-acc));
  }
}

// ---------------- K4c: pack scan operands to bf16 ----------------
// s=0 (yl): x=xr(raw), B=Br(raw), C=Cl(conv);  s=1 (yr): x=xl(conv), B=Bl(conv), C=Cr(raw)
__global__ void pack_bf16_kernel(const float* __restrict__ zxr,
                                 const float* __restrict__ xconv,
                                 __bf16* __restrict__ xbf, __bf16* __restrict__ Bbf,
                                 __bf16* __restrict__ Cbf) {
  int m = blockIdx.x, s = blockIdx.y;
  for (int c = threadIdx.x; c < CONVDIM; c += 256) {
    float v;
    if (s == 0) {
      if (c < 576)      v = zxr[(size_t)m * DPROJ + DINNER + c];   // xr, Br
      else              v = xconv[(size_t)m * CONVDIM + c];        // Cl
    } else {
      if (c < 576)      v = xconv[(size_t)m * CONVDIM + c];        // xl, Bl
      else              v = zxr[(size_t)m * DPROJ + DINNER + c];   // Cr
    }
    __bf16 b = (__bf16)v;
    if (c < 512)      xbf[((size_t)s * MTOK + m) * DINNER + c] = b;
    else if (c < 576) Bbf[((size_t)s * MTOK + m) * DSTATE + (c - 512)] = b;
    else              Cbf[((size_t)s * MTOK + m) * DSTATE + (c - 576)] = b;
  }
}

// ---------------- K5: per-chunk cumsum of dt*A ----------------
__global__ void cumsum_kernel(const float* __restrict__ dt2,
                              const float* __restrict__ A_log,
                              float* __restrict__ cum) {
  int c = blockIdx.x, bi = blockIdx.y, s = blockIdx.z;
  int i = threadIdx.x;  // 128
  __shared__ float buf[128];
  int m = bi * L + c * CHUNK + i;
  for (int h = 0; h < NH; ++h) {
    float Ah = -__expf(A_log[h]);
    buf[i] = dt2[((size_t)s * MTOK + m) * NH + h] * Ah;
    __syncthreads();
    for (int off = 1; off < 128; off <<= 1) {
      float add = (i >= off) ? buf[i - off] : 0.f;
      __syncthreads();
      buf[i] += add;
      __syncthreads();
    }
    cum[((((size_t)(s * BATCH + bi) * NC + c) * NH + h) << 7) + i] = buf[i];
    __syncthreads();
  }
}

// ---------------- K6: per-chunk intra work (CB, scores@x, states) ------------
__global__ __launch_bounds__(256)
void chunk_intra_kernel(const __bf16* __restrict__ xbf, const __bf16* __restrict__ Bbf,
                        const __bf16* __restrict__ Cbf, const float* __restrict__ dt2,
                        const float* __restrict__ cum, const float* __restrict__ Dh,
                        float* __restrict__ yf, float* __restrict__ states) {
  int c = blockIdx.x, bi = blockIdx.y, s = blockIdx.z;
  int t = threadIdx.x, lane = t & 31, wid = t >> 5;
  int m0 = bi * L + c * CHUNK;

  extern __shared__ char smem_raw[];
  __bf16* Bt    = (__bf16*)smem_raw;                 // [128][64]  16KB
  __bf16* scbuf = (__bf16*)(smem_raw + 16384);       // [128][128] 32KB (Ct alias first)
  __bf16* xT    = (__bf16*)(smem_raw + 49152);       // [128p][128j] 32KB
  __bf16* BwT   = (__bf16*)(smem_raw + 81920);       // [64n][128j] 16KB
  __shared__ float cumsh[NH][128];
  __shared__ float dtsh[NH][128];
  __shared__ float wsh[NH][128];

  __bf16* Ct = scbuf;
  {  // async-stage B,C tiles: 128x64 bf16 each
    int row = t >> 1, cs = (t & 1) * 32;
    const __bf16* bs = Bbf + ((size_t)s * MTOK + m0 + row) * DSTATE + cs;
    const __bf16* csrc = Cbf + ((size_t)s * MTOK + m0 + row) * DSTATE + cs;
    unsigned lb = lds_off_of(Bt + row * 64 + cs);
    unsigned lc = lds_off_of(Ct + row * 64 + cs);
#pragma unroll
    for (int q = 0; q < 4; ++q) {
      async_b128(lb + q * 16, bs + q * 8);
      async_b128(lc + q * 16, csrc + q * 8);
    }
  }
  for (int idx = t; idx < NH * 128; idx += 256) {
    int h = idx >> 7, i = idx & 127;
    cumsh[h][i] = cum[((((size_t)(s * BATCH + bi) * NC + c) * NH + h) << 7) + i];
    dtsh[h][i] = dt2[((size_t)s * MTOK + m0 + i) * NH + h];
  }
  wait_async0();
  __syncthreads();
  for (int idx = t; idx < NH * 128; idx += 256) {
    int h = idx >> 7, i = idx & 127;
    wsh[h][i] = dtsh[h][i] * __expf(cumsh[h][127] - cumsh[h][i]);
  }
  __syncthreads();

  // CB[i][j] = sum_n C[i][n] * B[j][n]   (M=128, N=128, K=64)
  int wm = wid >> 1, wn = wid & 1;  // 32-row x 64-col wave region
  v8f cb[2][4] = {};
  for (int kb = 0; kb < 64; kb += 32) {
    v16bf a0 = frag_a(Ct + (wm * 32) * 64 + kb, 64, lane);
    v16bf a1 = frag_a(Ct + (wm * 32 + 16) * 64 + kb, 64, lane);
#pragma unroll
    for (int j2 = 0; j2 < 4; ++j2) {
      v16bf b = frag_b(Bt + (wn * 64 + j2 * 16) * 64 + kb, 64, lane);
      cb[0][j2] = wmma_bf16(a0, b, cb[0][j2]);
      cb[1][j2] = wmma_bf16(a1, b, cb[1][j2]);
    }
  }
  __syncthreads();  // Ct dead; scbuf becomes scores

  for (int h = 0; h < NH; ++h) {
    // scores = tril(CB * exp(cum_i - cum_j)) * dt_j  -> bf16 into scbuf
    for (int i2 = 0; i2 < 2; ++i2)
      for (int j2 = 0; j2 < 4; ++j2) {
        int ib = wm * 32 + i2 * 16 + (lane >> 4) * 8;
        int j = wn * 64 + j2 * 16 + (lane & 15);
#pragma unroll
        for (int r = 0; r < 8; ++r) {
          int i = ib + r;
          float v = 0.f;
          if (i >= j) v = cb[i2][j2][r] * __expf(cumsh[h][i] - cumsh[h][j]) * dtsh[h][j];
          scbuf[i * 128 + j] = (__bf16)v;
        }
      }
    // x tile transposed: xT[p][j] = x[j][p]
    for (int idx = t; idx < 128 * 128; idx += 256) {
      int j = idx >> 7, p = idx & 127;
      xT[p * 128 + j] = xbf[((size_t)s * MTOK + m0 + j) * DINNER + h * HEADDIM + p];
    }
    // BwT[n][j] = B[j][n] * dt_j * exp(cum_last - cum_j)
    for (int idx = t; idx < 64 * 128; idx += 256) {
      int n = idx >> 7, j = idx & 127;
      BwT[n * 128 + j] = (__bf16)((float)Bt[j * 64 + n] * wsh[h][j]);
    }
    __syncthreads();

    // y_intra = scores @ x : (128 x 128) x (128 x 128)
    v8f ya[2][4] = {};
    for (int kb = 0; kb < 128; kb += 32) {
      v16bf a0 = frag_a(scbuf + (wm * 32) * 128 + kb, 128, lane);
      v16bf a1 = frag_a(scbuf + (wm * 32 + 16) * 128 + kb, 128, lane);
#pragma unroll
      for (int j2 = 0; j2 < 4; ++j2) {
        v16bf b = frag_b(xT + (wn * 64 + j2 * 16) * 128 + kb, 128, lane);
        ya[0][j2] = wmma_bf16(a0, b, ya[0][j2]);
        ya[1][j2] = wmma_bf16(a1, b, ya[1][j2]);
      }
    }
    float Dv = Dh[h];
    for (int i2 = 0; i2 < 2; ++i2)
      for (int j2 = 0; j2 < 4; ++j2) {
        int i = wm * 32 + i2 * 16 + (lane >> 4) * 8;
        int p = wn * 64 + j2 * 16 + (lane & 15);
#pragma unroll
        for (int r = 0; r < 8; ++r) {
          float xv = (float)xT[p * 128 + (i + r)];
          yf[((size_t)s * MTOK + m0 + i + r) * DINNER + h * HEADDIM + p] =
              ya[i2][j2][r] + Dv * xv;
        }
      }

    // states[n][p] = sum_j Bw[j][n] * x[j][p]  (M=64, N=128, K=128)
    v8f st[4] = {};
    for (int kb = 0; kb < 128; kb += 32) {
      v16bf a0 = frag_a(BwT + (wm * 16) * 128 + kb, 128, lane);
#pragma unroll
      for (int j2 = 0; j2 < 4; ++j2) {
        v16bf b = frag_b(xT + (wn * 64 + j2 * 16) * 128 + kb, 128, lane);
        st[j2] = wmma_bf16(a0, b, st[j2]);
      }
    }
    float* sbase = states + (((size_t)(s * BATCH + bi) * NC + c) * NH + h) * (DSTATE * HEADDIM);
    for (int j2 = 0; j2 < 4; ++j2) {
      int n = wm * 16 + (lane >> 4) * 8;
      int p = wn * 64 + j2 * 16 + (lane & 15);
#pragma unroll
      for (int r = 0; r < 8; ++r) sbase[(n + r) * HEADDIM + p] = st[j2][r];
    }
    __syncthreads();
  }
}

// ---------------- K7: sequential inter-chunk state scan ----------------
__global__ void state_scan_kernel(const float* __restrict__ states,
                                  const float* __restrict__ cum,
                                  float* __restrict__ hprev) {
  int id = blockIdx.x;                // s*16 + bi*4 + h
  int h = id & 3, bi = (id >> 2) & 3, s = id >> 4;
  int t = threadIdx.x;                // 256
  float hreg[32];
#pragma unroll
  for (int k = 0; k < 32; ++k) hreg[k] = 0.f;
  for (int c = 0; c < NC; ++c) {
    size_t base = (((size_t)(s * BATCH + bi) * NC + c) * NH + h) * (DSTATE * HEADDIM);
    float dec = __expf(cum[((((size_t)(s * BATCH + bi) * NC + c) * NH + h) << 7) + 127]);
#pragma unroll
    for (int k = 0; k < 32; ++k) {
      size_t idx = base + t + k * 256;
      hprev[idx] = hreg[k];
      hreg[k] = hreg[k] * dec + states[idx];
    }
  }
}

// ---------------- K8a: y += exp(cum_i) * (C @ hprev) ----------------
__global__ __launch_bounds__(256)
void chunk_inter_kernel(const __bf16* __restrict__ Cbf, const float* __restrict__ hprev,
                        const float* __restrict__ cum, float* __restrict__ yf) {
  int c = blockIdx.x, bi = blockIdx.y, s = blockIdx.z;
  int t = threadIdx.x, lane = t & 31, wid = t >> 5;
  int m0 = bi * L + c * CHUNK;
  __shared__ __bf16 Ct[128 * 64];
  __shared__ __bf16 hT[128 * 64];  // [p][n]
  __shared__ float cumsh[NH][128];
  int wm = wid >> 1, wn = wid & 1;
  {  // async-stage C tile
    int row = t >> 1, cs = (t & 1) * 32;
    const __bf16* csrc = Cbf + ((size_t)s * MTOK + m0 + row) * DSTATE + cs;
    unsigned lc = lds_off_of(Ct + row * 64 + cs);
#pragma unroll
    for (int q = 0; q < 4; ++q) async_b128(lc + q * 16, csrc + q * 8);
  }
  for (int idx = t; idx < NH * 128; idx += 256) {
    int h = idx >> 7, i = idx & 127;
    cumsh[h][i] = cum[((((size_t)(s * BATCH + bi) * NC + c) * NH + h) << 7) + i];
  }
  wait_async0();
  __syncthreads();
  for (int h = 0; h < NH; ++h) {
    const float* hb = hprev + (((size_t)(s * BATCH + bi) * NC + c) * NH + h) * (DSTATE * HEADDIM);
    for (int idx = t; idx < DSTATE * HEADDIM; idx += 256) {
      int n = idx >> 7, p = idx & 127;
      hT[p * 64 + n] = (__bf16)hb[n * HEADDIM + p];
    }
    __syncthreads();
    v8f acc[2][4] = {};
    for (int kb = 0; kb < 64; kb += 32) {
      v16bf a0 = frag_a(Ct + (wm * 32) * 64 + kb, 64, lane);
      v16bf a1 = frag_a(Ct + (wm * 32 + 16) * 64 + kb, 64, lane);
#pragma unroll
      for (int j2 = 0; j2 < 4; ++j2) {
        v16bf b = frag_b(hT + (wn * 64 + j2 * 16) * 64 + kb, 64, lane);
        acc[0][j2] = wmma_bf16(a0, b, acc[0][j2]);
        acc[1][j2] = wmma_bf16(a1, b, acc[1][j2]);
      }
    }
    for (int i2 = 0; i2 < 2; ++i2)
      for (int j2 = 0; j2 < 4; ++j2) {
        int i = wm * 32 + i2 * 16 + (lane >> 4) * 8;
        int p = wn * 64 + j2 * 16 + (lane & 15);
#pragma unroll
        for (int r = 0; r < 8; ++r) {
          size_t off = ((size_t)s * MTOK + m0 + i + r) * DINNER + h * HEADDIM + p;
          yf[off] += __expf(cumsh[h][i + r]) * acc[i2][j2][r];
        }
      }
    __syncthreads();
  }
}

// ---------------- K8b: gated RMSNorm -> bf16 ----------------
__global__ __launch_bounds__(256)
void gate_norm_kernel(const float* __restrict__ yf, const float* __restrict__ zxl,
                      const float* __restrict__ zxr, const float* __restrict__ norm_w,
                      __bf16* __restrict__ ygbf) {
  int m = blockIdx.x, s = blockIdx.y, t = threadIdx.x;
  const float* z = (s == 0 ? zxl : zxr) + (size_t)m * DPROJ;
  const float* y = yf + ((size_t)s * MTOK + m) * DINNER;
  __shared__ float red[256];
  float z0 = z[t], z1 = z[t + 256];
  float g0 = y[t] * (z0 / (1.f + __expf(-z0)));
  float g1 = y[t + 256] * (z1 / (1.f + __expf(-z1)));
  red[t] = g0 * g0 + g1 * g1;
  __syncthreads();
  for (int off = 128; off > 0; off >>= 1) {
    if (t < off) red[t] += red[t + off];
    __syncthreads();
  }
  float scale = rsqrtf(red[0] / (float)DINNER + 1e-5f);
  __bf16* o = ygbf + ((size_t)s * MTOK + m) * DINNER;
  o[t] = (__bf16)(g0 * scale * norm_w[t]);
  o[t + 256] = (__bf16)(g1 * scale * norm_w[t + 256]);
}

// ---------------- launcher ----------------
extern "C" void kernel_launch(void* const* d_in, const int* in_sizes, int n_in,
                              void* d_out, int out_size, void* d_ws, size_t ws_size,
                              hipStream_t stream) {
  const float* left   = (const float*)d_in[0];
  const float* right  = (const float*)d_in[1];
  const float* dw     = (const float*)d_in[2];
  const float* db     = (const float*)d_in[3];
  const float* winp   = (const float*)d_in[4];
  const float* convw  = (const float*)d_in[5];
  const float* convb  = (const float*)d_in[6];
  const float* dtbias = (const float*)d_in[7];
  const float* alog   = (const float*)d_in[8];
  const float* Dp     = (const float*)d_in[9];
  const float* normw  = (const float*)d_in[10];
  const float* woutp  = (const float*)d_in[11];
  float* out = (float*)d_out;

  char* ws = (char*)d_ws;
  size_t off = 0;
  auto take = [&](size_t bytes) -> char* {
    char* p = ws + off;
    off += (bytes + 255) & ~(size_t)255;
    return p;
  };
  __bf16* lfbf   = (__bf16*)take((size_t)MTOK * DMODEL * 2);
  __bf16* rfbf   = (__bf16*)take((size_t)MTOK * DMODEL * 2);
  __bf16* winbf  = (__bf16*)take((size_t)NPAD * DMODEL * 2);
  __bf16* woutbf = (__bf16*)take((size_t)DMODEL * DINNER * 2);
  float*  zxl    = (float*)take((size_t)MTOK * DPROJ * 4);
  float*  zxr    = (float*)take((size_t)MTOK * DPROJ * 4);
  float*  xconv  = (float*)take((size_t)MTOK * CONVDIM * 4);
  float*  dt2    = (float*)take((size_t)2 * MTOK * NH * 4);
  float*  cum    = (float*)take((size_t)2 * BATCH * NC * NH * CHUNK * 4);
  __bf16* xbf    = (__bf16*)take((size_t)2 * MTOK * DINNER * 2);
  __bf16* Bbf    = (__bf16*)take((size_t)2 * MTOK * DSTATE * 2);
  __bf16* Cbf    = (__bf16*)take((size_t)2 * MTOK * DSTATE * 2);
  float*  yf     = (float*)take((size_t)2 * MTOK * DINNER * 4);
  float*  states = (float*)take((size_t)2 * BATCH * NC * NH * DSTATE * HEADDIM * 4);
  float*  hprev  = (float*)take((size_t)2 * BATCH * NC * NH * DSTATE * HEADDIM * 4);
  __bf16* ygbf   = (__bf16*)take((size_t)2 * MTOK * DINNER * 2);

  // 1. downsample both images to bf16 tokens
  downsample_kernel<<<dim3(MTOK, 2), 256, 0, stream>>>(left, right, dw, db, lfbf, rfbf);
  // 2. weights to bf16
  pack_weights_kernel<<<(NPAD * DMODEL + 255) / 256, 256, 0, stream>>>(winp, woutp,
                                                                       winbf, woutbf);
  // 3. in_proj GEMMs (M=32768, N=1156 pad 1280, K=256)
  gemm_tn_kernel<<<dim3(MTOK / 128, NPAD / 128), 256, 0, stream>>>(
      lfbf, DMODEL, winbf, DMODEL, zxl, DPROJ, DMODEL, DPROJ);
  gemm_tn_kernel<<<dim3(MTOK / 128, NPAD / 128), 256, 0, stream>>>(
      rfbf, DMODEL, winbf, DMODEL, zxr, DPROJ, DMODEL, DPROJ);
  // 4. dt softplus, depthwise conv+SiLU (left), pack scan operands
  dt_kernel<<<(2 * MTOK * NH + 255) / 256, 256, 0, stream>>>(zxl, zxr, dtbias, dt2);
  conv_silu_kernel<<<MTOK, 256, 0, stream>>>(zxl, convw, convb, xconv);
  pack_bf16_kernel<<<dim3(MTOK, 2), 256, 0, stream>>>(zxr, xconv, xbf, Bbf, Cbf);
  // 5. per-chunk cumsum of dt*A
  cumsum_kernel<<<dim3(NC, BATCH, 2), 128, 0, stream>>>(dt2, alog, cum);
  // 6. intra-chunk: CB, scores@x (+ D*x), chunk states  [WMMA heavy]
  chunk_intra_kernel<<<dim3(NC, BATCH, 2), 256, 98304, stream>>>(xbf, Bbf, Cbf, dt2, cum,
                                                                 Dp, yf, states);
  // 7. sequential inter-chunk recurrence
  state_scan_kernel<<<32, 256, 0, stream>>>(states, cum, hprev);
  // 8. inter-chunk contribution: y += exp(cum) * C @ hprev
  chunk_inter_kernel<<<dim3(NC, BATCH, 2), 256, 0, stream>>>(Cbf, hprev, cum, yf);
  // 9. gated RMSNorm -> bf16
  gate_norm_kernel<<<dim3(MTOK, 2), 256, 0, stream>>>(yf, zxl, zxr, normw, ygbf);
  // 10. out_proj GEMMs into d_out (out_l then out_r, each (b,h,w,c) = (m, c))
  gemm_tn_kernel<<<dim3(MTOK / 128, DMODEL / 128), 256, 0, stream>>>(
      ygbf, DINNER, woutbf, DINNER, out, DMODEL, DINNER, DMODEL);
  gemm_tn_kernel<<<dim3(MTOK / 128, DMODEL / 128), 256, 0, stream>>>(
      ygbf + (size_t)MTOK * DINNER, DINNER, woutbf, DINNER,
      out + (size_t)MTOK * DMODEL, DMODEL, DINNER, DMODEL);
}
